// MultiheadAttention_3805341024705
// MI455X (gfx1250) — matrix operand order repaired and checked by reading, not compiled, not run on hardware
//
#include <hip/hip_runtime.h>
#include <hip/hip_bf16.h>

typedef __attribute__((ext_vector_type(16))) __bf16 v16bf;
typedef __attribute__((ext_vector_type(8)))  __bf16 v8bf;
typedef __attribute__((ext_vector_type(8)))  float  v8f;

#define T_DIM 1024
#define B_DIM 4
#define E_DIM 768
#define H_DIM 12
#define D_DIM 64
#define BH_DIM (B_DIM * H_DIM)   // 48
#define ROWS (T_DIM * B_DIM)     // 4096

// ---------- bf16 helpers ----------
static __device__ __forceinline__ unsigned short f2bf(float f) {
  __bf16 h = (__bf16)f;
  return __builtin_bit_cast(unsigned short, h);
}

// concat two 8-element bf16 loads into a 16-element WMMA operand
static __device__ __forceinline__ v16bf ldpair(const unsigned short* p0,
                                               const unsigned short* p1) {
  v8bf lo = *(const v8bf*)p0;
  v8bf hi = *(const v8bf*)p1;
  return __builtin_shufflevector(lo, hi, 0,1,2,3,4,5,6,7,8,9,10,11,12,13,14,15);
}

// A-matrix 16x32 bf16 layout: lane<16 -> K = kb+{0..7}, kb+16+{0..7} with kb=0
//                             lane>=16 -> kb=8  (ISA 7.12.2 table)
static __device__ __forceinline__ v16bf ldA_bf16(const unsigned short* row, int kb) {
  return ldpair(row + kb, row + kb + 16);
}
// B-matrix 32x16 bf16 layout: same as A: lane<16 -> K=0..7,16..23 ; lane>=16 -> K=8..15,24..31
static __device__ __forceinline__ v16bf ldB_bf16(const unsigned short* row, int koff) {
  return ldpair(row + koff, row + koff + 16);
}

// ---------- kernel 0: one-pass f32 -> bf16 convert (8 elems / thread) ----------
__global__ __launch_bounds__(256) void cvt_bf16(const float* __restrict__ src,
                                                unsigned short* __restrict__ dst,
                                                int n8) {
  int i = blockIdx.x * 256 + threadIdx.x;
  if (i >= n8) return;
  const float4* s = (const float4*)src;
  float4 a = s[2 * i], b = s[2 * i + 1];
  v8bf r;
  r[0] = (__bf16)a.x; r[1] = (__bf16)a.y; r[2] = (__bf16)a.z; r[3] = (__bf16)a.w;
  r[4] = (__bf16)b.x; r[5] = (__bf16)b.y; r[6] = (__bf16)b.z; r[7] = (__bf16)b.w;
  ((v8bf*)dst)[i] = r;
}

// ---------- kernel 1/3: Y = X[4096,768] @ W^T + bias (X, W already bf16) ----------
// mode 0: bf16 out[bh][t][64] * scale   (q / k)
// mode 1: bf16 out[bh][64][t]           (v, transposed for P@V B-operand)
// mode 2: f32  out[r][768]              (final output projection)
__global__ __launch_bounds__(128) void proj_gemm(const unsigned short* __restrict__ X,
                                                 const unsigned short* __restrict__ W,
                                                 const float* __restrict__ bias,
                                                 void* __restrict__ out,
                                                 float scale, int mode) {
  const int lane = threadIdx.x & 31;
  const int wave = threadIdx.x >> 5;
  const int tile = blockIdx.x * 4 + wave;          // 12288 tiles total
  const int mt = tile / (E_DIM / 16);
  const int nt = tile % (E_DIM / 16);
  const int kb   = (lane < 16) ? 0 : 8;
  const int koff = (lane < 16) ? 0 : 8;
  const int ln = lane & 15;
  const int rowbase = (lane >= 16) ? 8 : 0;

  const int r = mt * 16 + ln;                      // X row for this lane
  const int f = nt * 16 + ln;                      // output feature / W row
  const unsigned short* xrow = X + (size_t)r * E_DIM;
  const unsigned short* wrow = W + (size_t)f * E_DIM;

  v8f acc = {};
#pragma unroll 6
  for (int k0 = 0; k0 < E_DIM; k0 += 32) {
    v16bf a = ldA_bf16(xrow + k0, kb);
    v16bf b = ldB_bf16(wrow + k0, koff);
    acc = __builtin_amdgcn_wmma_f32_16x16x32_bf16(false, a, false, b,
                                                  (short)0, acc, false, false);
  }

  const float bv = bias[f];
  if (mode == 2) {
    float* o = (float*)out;
#pragma unroll
    for (int i = 0; i < 8; ++i) {
      int m = mt * 16 + rowbase + i;
      o[(size_t)m * E_DIM + f] = acc[i] + bv;
    }
  } else {
    unsigned short* o = (unsigned short*)out;
    const int h = f >> 6, d = f & 63;
#pragma unroll
    for (int i = 0; i < 8; ++i) {
      int m = mt * 16 + rowbase + i;                // m = t*B + b
      int t = m >> 2, bb = m & 3;
      int bh = bb * H_DIM + h;
      unsigned short val = f2bf((acc[i] + bv) * scale);
      if (mode == 0) o[((size_t)bh * T_DIM + t) * D_DIM + d] = val;
      else           o[((size_t)bh * D_DIM + d) * T_DIM + t] = val;
    }
  }
}

// ---------- kernel 2/3: fused scores + softmax + P@V + head-average ----------
// one block (4 waves) per (bh, 16-row t tile); wave w owns s in [w*256,(w+1)*256)
__global__ __launch_bounds__(128) void attn_kernel(const unsigned short* __restrict__ q,
                                                   const unsigned short* __restrict__ k,
                                                   const unsigned short* __restrict__ vt,
                                                   const float* __restrict__ bias,
                                                   unsigned short* __restrict__ attn_out, // [T,B,E] bf16
                                                   float* __restrict__ avg) {             // [B,T,T] f32
  __shared__ unsigned short Pl[16][1032];   // probs (bf16), row padded 16B vs banks
  __shared__ float redmax[4][16];
  __shared__ float redsum[4][16];

  const int lane = threadIdx.x & 31;
  const int wave = threadIdx.x >> 5;
  const int bh = blockIdx.x >> 6;
  const int tt = blockIdx.x & 63;
  const int bidx = bh / H_DIM;
  const int h = bh % H_DIM;
  const int kb   = (lane < 16) ? 0 : 8;
  const int koff = (lane < 16) ? 0 : 8;
  const int ln = lane & 15;
  const int rowbase = (lane >= 16) ? 8 : 0;

  // q tile A operands (16x64 -> two K=32 chunks), reused across all 16 s-tiles
  const unsigned short* qrow = q + ((size_t)bh * T_DIM + tt * 16 + ln) * D_DIM;
  v16bf a0 = ldA_bf16(qrow, kb);
  v16bf a1 = ldA_bf16(qrow + 32, kb);

  const int sbase = wave * 256;
  v8f acc[16];

  // ---- scores: S = q @ k^T + bias, accumulators seeded with bias ----
#pragma unroll
  for (int j = 0; j < 16; ++j) {
    const int scol = sbase + j * 16 + ln;
    const float* bptr = bias + ((size_t)bh * T_DIM + (tt * 16 + rowbase)) * T_DIM + scol;
    v8f c;
#pragma unroll
    for (int i = 0; i < 8; ++i) c[i] = bptr[(size_t)i * T_DIM];
    const unsigned short* krow = k + ((size_t)bh * T_DIM + scol) * D_DIM;
    v16bf b0 = ldB_bf16(krow, koff);
    v16bf b1 = ldB_bf16(krow + 32, koff);
    c = __builtin_amdgcn_wmma_f32_16x16x32_bf16(false, a0, false, b0, (short)0, c, false, false);
    c = __builtin_amdgcn_wmma_f32_16x16x32_bf16(false, a1, false, b1, (short)0, c, false, false);
    acc[j] = c;
  }

  // ---- softmax (row max / sum across 4 waves) ----
  float gm[8], rs[8];
#pragma unroll
  for (int i = 0; i < 8; ++i) {
    float m = -3.0e38f;
#pragma unroll
    for (int j = 0; j < 16; ++j) m = fmaxf(m, acc[j][i]);
    for (int off = 1; off < 16; off <<= 1) m = fmaxf(m, __shfl_xor(m, off, 32));
    gm[i] = m;
  }
  if (ln == 0) {
#pragma unroll
    for (int i = 0; i < 8; ++i) redmax[wave][rowbase + i] = gm[i];
  }
  __syncthreads();
#pragma unroll
  for (int i = 0; i < 8; ++i) {
    float m = fmaxf(fmaxf(redmax[0][rowbase + i], redmax[1][rowbase + i]),
                    fmaxf(redmax[2][rowbase + i], redmax[3][rowbase + i]));
    gm[i] = m;
  }
#pragma unroll
  for (int i = 0; i < 8; ++i) {
    float s = 0.f;
#pragma unroll
    for (int j = 0; j < 16; ++j) {
      float e = __expf(acc[j][i] - gm[i]);
      acc[j][i] = e;
      s += e;
    }
    for (int off = 1; off < 16; off <<= 1) s += __shfl_xor(s, off, 32);
    rs[i] = s;
  }
  if (ln == 0) {
#pragma unroll
    for (int i = 0; i < 8; ++i) redsum[wave][rowbase + i] = rs[i];
  }
  __syncthreads();
#pragma unroll
  for (int i = 0; i < 8; ++i) {
    float s = redsum[0][rowbase + i] + redsum[1][rowbase + i] +
              redsum[2][rowbase + i] + redsum[3][rowbase + i];
    rs[i] = 1.0f / s;
  }

  // ---- probs -> LDS (bf16) + head-average accumulation (atomic f32) ----
#pragma unroll
  for (int j = 0; j < 16; ++j) {
    const int scol = sbase + j * 16 + ln;
#pragma unroll
    for (int i = 0; i < 8; ++i) {
      float p = acc[j][i] * rs[i];
      Pl[rowbase + i][scol] = f2bf(p);
      atomicAdd(&avg[((size_t)bidx * T_DIM + tt * 16 + rowbase + i) * T_DIM + scol],
                p * (1.0f / H_DIM));
    }
  }
  __syncthreads();

  // ---- O = P @ V : wave owns d-columns [wave*16, wave*16+16), K = s = 1024 ----
  const unsigned short* vrow = vt + ((size_t)bh * D_DIM + wave * 16 + ln) * T_DIM;
  v8f o = {};
#pragma unroll 8
  for (int c = 0; c < 32; ++c) {
    const unsigned short* prow = &Pl[ln][c * 32];
    v16bf a = ldA_bf16(prow, kb);
    v16bf b = ldB_bf16(vrow + c * 32, koff);
    o = __builtin_amdgcn_wmma_f32_16x16x32_bf16(false, a, false, b, (short)0, o, false, false);
  }
#pragma unroll
  for (int i = 0; i < 8; ++i) {
    int tg = tt * 16 + rowbase + i;
    int e = h * 64 + wave * 16 + ln;
    attn_out[((size_t)tg * B_DIM + bidx) * E_DIM + e] = f2bf(o[i]);
  }
}

// ---------- host ----------
extern "C" void kernel_launch(void* const* d_in, const int* in_sizes, int n_in,
                              void* d_out, int out_size, void* d_ws, size_t ws_size,
                              hipStream_t stream) {
  (void)in_sizes; (void)n_in; (void)out_size; (void)ws_size;
  const float* query = (const float*)d_in[0];
  const float* attn_bias = (const float*)d_in[1];
  const float* Wq = (const float*)d_in[2]; const float* bq = (const float*)d_in[3];
  const float* Wk = (const float*)d_in[4]; const float* bk = (const float*)d_in[5];
  const float* Wv = (const float*)d_in[6]; const float* bv = (const float*)d_in[7];
  const float* Wo = (const float*)d_in[8]; const float* bo = (const float*)d_in[9];

  float* out = (float*)d_out;                              // [T,B,E]
  float* avg = out + (size_t)T_DIM * B_DIM * E_DIM;        // [B,T,T]

  char* ws = (char*)d_ws;
  const size_t bf16_head = (size_t)BH_DIM * T_DIM * D_DIM * sizeof(unsigned short);
  const size_t bf16_x    = (size_t)ROWS * E_DIM * sizeof(unsigned short);
  const size_t bf16_w    = (size_t)E_DIM * E_DIM * sizeof(unsigned short);
  unsigned short* qws = (unsigned short*)ws; ws += bf16_head;
  unsigned short* kws = (unsigned short*)ws; ws += bf16_head;
  unsigned short* vws = (unsigned short*)ws; ws += bf16_head;
  unsigned short* xbf = (unsigned short*)ws; ws += bf16_x;   // bf16 query [4096,768]
  unsigned short* abf = (unsigned short*)ws; ws += bf16_x;   // bf16 attn  [T,B,E]
  unsigned short* wqb = (unsigned short*)ws; ws += bf16_w;
  unsigned short* wkb = (unsigned short*)ws; ws += bf16_w;
  unsigned short* wvb = (unsigned short*)ws; ws += bf16_w;
  unsigned short* wob = (unsigned short*)ws; ws += bf16_w;

  hipMemsetAsync(avg, 0, (size_t)B_DIM * T_DIM * T_DIM * sizeof(float), stream);

  // one-pass f32 -> bf16 conversion of activations and weights
  {
    const int nx8 = ROWS * E_DIM / 8;       // 393216
    const int nw8 = E_DIM * E_DIM / 8;      // 73728
    cvt_bf16<<<dim3((nx8 + 255) / 256), dim3(256), 0, stream>>>(query, xbf, nx8);
    cvt_bf16<<<dim3((nw8 + 255) / 256), dim3(256), 0, stream>>>(Wq, wqb, nw8);
    cvt_bf16<<<dim3((nw8 + 255) / 256), dim3(256), 0, stream>>>(Wk, wkb, nw8);
    cvt_bf16<<<dim3((nw8 + 255) / 256), dim3(256), 0, stream>>>(Wv, wvb, nw8);
    cvt_bf16<<<dim3((nw8 + 255) / 256), dim3(256), 0, stream>>>(Wo, wob, nw8);
  }

  dim3 blk(128);
  dim3 grd((ROWS / 16) * (E_DIM / 16) / 4);                // 3072 blocks, 4 wave-tiles each
  const float scale = 0.125f;                              // D^-0.5 = 1/8

  proj_gemm<<<grd, blk, 0, stream>>>(xbf, wqb, bq, (void*)qws, scale, 0);
  proj_gemm<<<grd, blk, 0, stream>>>(xbf, wkb, bk, (void*)kws, 1.0f, 0);
  proj_gemm<<<grd, blk, 0, stream>>>(xbf, wvb, bv, (void*)vws, 1.0f, 1);

  attn_kernel<<<dim3(BH_DIM * (T_DIM / 16)), blk, 0, stream>>>(qws, kws, vws, attn_bias,
                                                               abf, avg);

  proj_gemm<<<grd, blk, 0, stream>>>(abf, wob, bo, (void*)out, 1.0f, 2);
}
